// VisionMamba_32538672235138
// MI455X (gfx1250) — matrix operand order repaired and checked by reading, not compiled, not run on hardware
//
#include <hip/hip_runtime.h>

// ---------------------------------------------------------------------------
// VisionMamba forward for MI455X (gfx1250, wave32, WMMA).
// GEMMs: v_wmma_f32_16x16x32_bf16, LDS fragments via 128-bit ds loads,
// weights stored N-major (transposed) so B fragments are contiguous.
// Weight tiles + scan B/C rows staged with GLOBAL_LOAD_ASYNC_TO_LDS (ASYNCcnt).
// ---------------------------------------------------------------------------

typedef __attribute__((ext_vector_type(16))) __bf16 v16bf;
typedef __attribute__((ext_vector_type(8)))  __bf16 v8bf;
typedef __attribute__((ext_vector_type(8)))  float  v8f;

#define NLAYERS 8
#define FDIM    64
#define NSTATE  64
#define NDELTA  4
#define LSEQ    1024
#define BSZ     4
#define ROWS    (BSZ * LSEQ)   // 4096
#define EPSV    1e-5f

__device__ __forceinline__ unsigned short bf16_bits(float f) {
  unsigned u = __float_as_uint(f);
  return (unsigned short)((u + 0x7FFFu + ((u >> 16) & 1u)) >> 16);
}
__device__ __forceinline__ __bf16 to_bf16(float f) {
  unsigned short us = bf16_bits(f);
  return __builtin_bit_cast(__bf16, us);
}
__device__ __forceinline__ unsigned pack2bf(float a, float b) {
  return (unsigned)bf16_bits(a) | ((unsigned)bf16_bits(b) << 16);
}
__device__ __forceinline__ float siluf_(float x) { return x / (1.f + __expf(-x)); }
__device__ __forceinline__ float softplusf_(float x) {
  return (x > 0.f) ? (x + log1pf(__expf(-x))) : log1pf(__expf(x));
}

// ---- CDNA5 async global->LDS (ASYNCcnt) --------------------------------
__device__ __forceinline__ unsigned lds_addr_of(const void* p) {
  return (unsigned)(uintptr_t)p;   // LDS aperture: offset lives in addr[31:0]
}
__device__ __forceinline__ void async_load_b128(unsigned lds, unsigned goff,
                                                const void* sbase) {
  asm volatile("global_load_async_to_lds_b128 %0, %1, %2"
               :: "v"(lds), "v"(goff), "s"(sbase) : "memory");
}
__device__ __forceinline__ void async_load_b32(unsigned lds, unsigned goff,
                                               const void* sbase) {
  asm volatile("global_load_async_to_lds_b32 %0, %1, %2"
               :: "v"(lds), "v"(goff), "s"(sbase) : "memory");
}
__device__ __forceinline__ void wait_async0() {
  asm volatile("s_wait_asynccnt 0x0" ::: "memory");
}
__device__ __forceinline__ void wait_async2() {
  asm volatile("s_wait_asynccnt 0x2" ::: "memory");
}

// A fragment (16-bit A 16x32 layout): two contiguous 8-elem runs -> 2x b128.
__device__ __forceinline__ v16bf load_a_frag(const __bf16* Abase, int kh, int kb) {
  v8bf lo = *(const v8bf*)(Abase + kb + kh);
  v8bf hi = *(const v8bf*)(Abase + kb + 16 + kh);
  return __builtin_shufflevector(lo, hi, 0, 1, 2, 3, 4, 5, 6, 7,
                                 8, 9, 10, 11, 12, 13, 14, 15);
}
// B fragment from N-major weights: two contiguous 8-elem runs (A layout) -> 2x b128.
__device__ __forceinline__ v16bf load_b_frag(const __bf16* Wt, int col, int ko, int kb) {
  v8bf lo = *(const v8bf*)(Wt + col * 64 + kb + (ko >> 1));
  v8bf hi = *(const v8bf*)(Wt + col * 64 + kb + 16 + (ko >> 1));
  return __builtin_shufflevector(lo, hi, 0, 1, 2, 3, 4, 5, 6, 7,
                                 8, 9, 10, 11, 12, 13, 14, 15);
}

// ---------------------------------------------------------------------------
// Weight conversion: f32 -> bf16, stored N-major (Wt[n][k]) for WMMA B frags.
// ---------------------------------------------------------------------------
__global__ void convert_weights_kernel(
    const float* __restrict__ Wp_in, const float* __restrict__ Wconv,
    const float* __restrict__ WB, const float* __restrict__ WC,
    const float* __restrict__ bB, const float* __restrict__ bC,
    const float* __restrict__ Wp_out,
    __bf16* __restrict__ wpin_b, __bf16* __restrict__ wconv_b,
    __bf16* __restrict__ wbc_b, float* __restrict__ bbc,
    __bf16* __restrict__ wpout_b)
{
  int i = blockIdx.x * 256 + threadIdx.x;
  if (i < NLAYERS * 64 * 64) {         // dst[l][n][k] = W[l][k][n]
    int l = i >> 12; int r = i & 4095;
    int n = r >> 6, k = r & 63;
    wpin_b[i]  = to_bf16(Wp_in[(l * 64 + k) * 64 + n]);
    wpout_b[i] = to_bf16(Wp_out[(l * 64 + k) * 64 + n]);
  }
  if (i < NLAYERS * 3 * 64 * 64) {     // dst[l][tap][o][c] = Wconv[l][o][c][tap]
    int l = i / (3 * 64 * 64); int r = i % (3 * 64 * 64);
    int tap = r / 4096; int r2 = r & 4095;
    int o = r2 >> 6, c = r2 & 63;
    wconv_b[i] = to_bf16(Wconv[((l * 64 + o) * 64 + c) * 3 + tap]);
  }
  if (i < NLAYERS * 128 * 64) {        // dst[l][n][k], n in [0,128)
    int l = i >> 13; int r = i & 8191;
    int n = r >> 6, k = r & 63;
    float v = (n < 64) ? WB[(l * 64 + k) * 64 + n] : WC[(l * 64 + k) * 64 + (n - 64)];
    wbc_b[i] = to_bf16(v);
  }
  if (i < NLAYERS * 128) {
    int l = i >> 7, n = i & 127;
    bbc[i] = (n < 64) ? bB[l * 64 + n] : bC[l * 64 + (n - 64)];
  }
}

// ---------------------------------------------------------------------------
// Patch embedding (Conv2d k=stride=8) + sinusoidal positional embedding.
// ---------------------------------------------------------------------------
__global__ void patch_embed_kernel(const float* __restrict__ x,
                                   const float* __restrict__ Wpe,
                                   const float* __restrict__ bpe,
                                   float* __restrict__ h)
{
  __shared__ __align__(16) float xs[192];
  int bl = blockIdx.x;
  int b = bl >> 10, l = bl & 1023;
  int i = l >> 5, j = l & 31;
  int t = threadIdx.x;   // 64
  for (int idx = t; idx < 192; idx += 64) {
    int c = idx >> 6; int pq = idx & 63; int p = pq >> 3, q = pq & 7;
    xs[idx] = x[((b * 3 + c) * 256 + i * 8 + p) * 256 + j * 8 + q];
  }
  __syncthreads();
  int f = t;
  float s = bpe[f];
  #pragma unroll
  for (int c = 0; c < 3; ++c)
    for (int pq = 0; pq < 64; ++pq)
      s += xs[c * 64 + pq] * Wpe[(f * 3 + c) * 64 + pq];
  int k = f >> 1;
  float ang = (float)l * expf((float)(2 * k) * (-9.210340372f / 64.f));
  s += (f & 1) ? cosf(ang) : sinf(ang);
  h[bl * 64 + f] = s;
}

// ---------------------------------------------------------------------------
// GroupNorm(1) stats over (L,F) per sample.
// ---------------------------------------------------------------------------
__global__ void gn_stats_kernel(const float* __restrict__ h, float* __restrict__ stats)
{
  int b = blockIdx.x, t = threadIdx.x;   // 256 threads
  float s = 0.f, ss = 0.f;
  for (int idx = t; idx < LSEQ * 16; idx += 256) {
    float4 v = ((const float4*)h)[b * LSEQ * 16 + idx];
    s += v.x + v.y + v.z + v.w;
    ss += v.x * v.x + v.y * v.y + v.z * v.z + v.w * v.w;
  }
  __shared__ float rs[256], rss[256];
  rs[t] = s; rss[t] = ss; __syncthreads();
  for (int o = 128; o > 0; o >>= 1) {
    if (t < o) { rs[t] += rs[t + o]; rss[t] += rss[t + o]; }
    __syncthreads();
  }
  if (t == 0) {
    float m = rs[0] / 65536.f;
    float v = rss[0] / 65536.f - m * m;
    stats[b * 2] = m; stats[b * 2 + 1] = rsqrtf(v + EPSV);
  }
}

// ---------------------------------------------------------------------------
// Generic GEMM: out[row, 0:N] = norm(X[row, 0:64]) @ W(64xN) + bias.
// Weights DMA'd to LDS asynchronously while the activation tile is converted.
// ---------------------------------------------------------------------------
__global__ void gemm_gen_kernel(const float* __restrict__ X,
                                const float* __restrict__ stats,   // per-b (mean,rstd) or null
                                const __bf16* __restrict__ W,      // N-major (N x 64)
                                const float* __restrict__ bias,
                                float* __restrict__ out, int N)
{
  __shared__ __align__(32) __bf16 ldsA[128 * 64];
  __shared__ __align__(32) __bf16 ldsW[128 * 64];
  const int t = threadIdx.x;
  const int r0 = blockIdx.x * 128;
  float mean = 0.f, rstd = 1.f;
  if (stats) { int b = r0 >> 10; mean = stats[b * 2]; rstd = stats[b * 2 + 1]; }
  const unsigned ldsw_base = lds_addr_of(ldsW);
  for (int idx = t; idx < N * 8; idx += 256)        // 16B chunks, async DMA
    async_load_b128(ldsw_base + (unsigned)idx * 16u, (unsigned)idx * 16u, W);
  for (int idx = t; idx < 128 * 16; idx += 256) {   // 4 floats -> 4 bf16
    float4 v = ((const float4*)X)[r0 * 16 + idx];
    uint2 pk;
    pk.x = pack2bf((v.x - mean) * rstd, (v.y - mean) * rstd);
    pk.y = pack2bf((v.z - mean) * rstd, (v.w - mean) * rstd);
    ((uint2*)ldsA)[idx] = pk;
  }
  wait_async0();
  __syncthreads();

  const int lane = t & 31, wave = t >> 5;
  const int m = lane & 15, kh = (lane >> 4) * 8;
  const int n = lane & 15, ko = (lane >> 4) * 16, mh = (lane >> 4) * 8;

  const __bf16* Abase = &ldsA[(wave * 16 + m) * 64];
  v16bf a0 = load_a_frag(Abase, kh, 0);
  v16bf a1 = load_a_frag(Abase, kh, 32);

  for (int nt = 0; nt < (N >> 4); ++nt) {
    v8f c;
    float bv = bias[nt * 16 + n];
    #pragma unroll
    for (int i = 0; i < 8; ++i) c[i] = bv;
    v16bf b0 = load_b_frag(ldsW, nt * 16 + n, ko, 0);
    v16bf b1 = load_b_frag(ldsW, nt * 16 + n, ko, 32);
    c = __builtin_amdgcn_wmma_f32_16x16x32_bf16(false, a0, false, b0, (short)0, c, false, false);
    c = __builtin_amdgcn_wmma_f32_16x16x32_bf16(false, a1, false, b1, (short)0, c, false, false);
    #pragma unroll
    for (int i = 0; i < 8; ++i)
      out[(r0 + wave * 16 + mh + i) * N + nt * 16 + n] = c[i];
  }
}

// ---------------------------------------------------------------------------
// Conv1d(64->64, k=3, pad=1) along L as 3 shifted GEMMs. LDS halo rows zeroed
// at sequence boundaries (tiles never cross a batch boundary: 1024 % 128 == 0).
// ---------------------------------------------------------------------------
__global__ void conv_gemm_kernel(const float* __restrict__ h0,
                                 const __bf16* __restrict__ Wc,  // [tap][o][c] N-major
                                 const float* __restrict__ bc,
                                 float* __restrict__ out)
{
  __shared__ __align__(32) __bf16 ldsA[130 * 64];
  __shared__ __align__(32) __bf16 ldsW[3 * 64 * 64];
  const int t = threadIdx.x;
  const int r0 = blockIdx.x * 128;
  const int b = r0 >> 10, l0 = r0 & 1023;
  const unsigned ldsw_base = lds_addr_of(ldsW);
  for (int idx = t; idx < 3 * 64 * 8; idx += 256)   // async DMA weights
    async_load_b128(ldsw_base + (unsigned)idx * 16u, (unsigned)idx * 16u, Wc);
  for (int idx = t; idx < 130 * 16; idx += 256) {
    int row = idx >> 4;
    int l = l0 + row - 1;
    uint2 pk; pk.x = 0u; pk.y = 0u;
    if (l >= 0 && l < LSEQ) {
      float4 v = ((const float4*)h0)[(((b << 10) + l) << 4) + (idx & 15)];
      pk.x = pack2bf(v.x, v.y);
      pk.y = pack2bf(v.z, v.w);
    }
    ((uint2*)ldsA)[idx] = pk;
  }
  wait_async0();
  __syncthreads();

  const int lane = t & 31, wave = t >> 5;
  const int m = lane & 15, kh = (lane >> 4) * 8;
  const int n = lane & 15, ko = (lane >> 4) * 16, mh = (lane >> 4) * 8;

  v8f acc[4];
  for (int nt = 0; nt < 4; ++nt) {
    float bv = bc[nt * 16 + n];
    #pragma unroll
    for (int i = 0; i < 8; ++i) acc[nt][i] = bv;
  }
  for (int tap = 0; tap < 3; ++tap) {
    const __bf16* Abase = &ldsA[(wave * 16 + m + tap) * 64];   // lds row 0 == l0-1
    v16bf a0 = load_a_frag(Abase, kh, 0);
    v16bf a1 = load_a_frag(Abase, kh, 32);
    const __bf16* Wbase = &ldsW[tap * 4096];
    for (int nt = 0; nt < 4; ++nt) {
      v16bf b0 = load_b_frag(Wbase, nt * 16 + n, ko, 0);
      v16bf b1 = load_b_frag(Wbase, nt * 16 + n, ko, 32);
      acc[nt] = __builtin_amdgcn_wmma_f32_16x16x32_bf16(false, a0, false, b0, (short)0, acc[nt], false, false);
      acc[nt] = __builtin_amdgcn_wmma_f32_16x16x32_bf16(false, a1, false, b1, (short)0, acc[nt], false, false);
    }
  }
  for (int nt = 0; nt < 4; ++nt) {
    #pragma unroll
    for (int i = 0; i < 8; ++i)
      out[(r0 + wave * 16 + mh + i) * 64 + nt * 16 + n] = acc[nt][i];
  }
}

// ---------------------------------------------------------------------------
// InstanceNorm over L per (b, channel): stats then apply (+SiLU).
// ---------------------------------------------------------------------------
__global__ void inorm_stats_kernel(const float* __restrict__ hcr, float* __restrict__ istats)
{
  int bf = blockIdx.x;            // b*64 + f
  int b = bf >> 6, f = bf & 63;
  int t = threadIdx.x;            // 256
  float s = 0.f, ss = 0.f;
  for (int l = t; l < LSEQ; l += 256) {
    float v = hcr[((b << 10) + l) * 64 + f];
    s += v; ss += v * v;
  }
  __shared__ float rs[256], rss[256];
  rs[t] = s; rss[t] = ss; __syncthreads();
  for (int o = 128; o > 0; o >>= 1) {
    if (t < o) { rs[t] += rs[t + o]; rss[t] += rss[t + o]; }
    __syncthreads();
  }
  if (t == 0) {
    float m = rs[0] / (float)LSEQ;
    float v = rss[0] / (float)LSEQ - m * m;
    istats[bf * 2] = m; istats[bf * 2 + 1] = rsqrtf(v + EPSV);
  }
}

__global__ void inorm_apply_kernel(const float* __restrict__ hcr,
                                   const float* __restrict__ istats,
                                   float* __restrict__ hc)
{
  int idx = blockIdx.x * 256 + threadIdx.x;
  if (idx >= ROWS * 64) return;
  int f = idx & 63;
  int b = idx >> 16;              // idx / (1024*64)
  int bf = (b << 6) + f;
  float xn = (hcr[idx] - istats[bf * 2]) * istats[bf * 2 + 1];
  hc[idx] = siluf_(xn);
}

// ---------------------------------------------------------------------------
// Delta = softplus((hc @ Wd1 + bd1) @ Wd2 + bd2). One 64-thread block per row.
// ---------------------------------------------------------------------------
__global__ void delta_kernel(const float* __restrict__ hc,
                             const float* __restrict__ Wd1, const float* __restrict__ bd1,
                             const float* __restrict__ Wd2, const float* __restrict__ bd2,
                             float* __restrict__ Del)
{
  __shared__ __align__(16) float srow[64];
  __shared__ float d1[NDELTA];
  int row = blockIdx.x, t = threadIdx.x;
  srow[t] = hc[row * 64 + t];
  __syncthreads();
  if (t < NDELTA) {
    float s = bd1[t];
    #pragma unroll
    for (int k = 0; k < 64; ++k) s += srow[k] * Wd1[k * NDELTA + t];
    d1[t] = s;
  }
  __syncthreads();
  float s = bd2[t];
  #pragma unroll
  for (int nd = 0; nd < NDELTA; ++nd) s += d1[nd] * Wd2[nd * 64 + t];
  Del[row * 64 + t] = softplusf_(s);
}

// ---------------------------------------------------------------------------
// Selective scan: thread = (b,f) with 64 states + 64 A values in VGPRs.
// B/C rows double-buffered in LDS via async DMA (row l+1 in flight while
// computing row l; async loads retire in order, 2 issued per thread per row,
// so s_wait_asynccnt<=2 guarantees the current row has landed).
// ---------------------------------------------------------------------------
__global__ void scan_kernel(const float* __restrict__ Del,
                            const float* __restrict__ hc,
                            const float* __restrict__ BmCm,
                            const float* __restrict__ A_log,   // layer slice (64,64)
                            const float* __restrict__ Dp,      // layer slice (64,)
                            float* __restrict__ y)
{
  int b = blockIdx.x;
  int f = threadIdx.x;            // 64 threads
  __shared__ __align__(16) float sBC[2][128];
  const unsigned lds_base = lds_addr_of(&sBC[0][0]);
  float a[NSTATE], s[NSTATE];
  #pragma unroll
  for (int n = 0; n < NSTATE; ++n) { a[n] = __expf(A_log[f * NSTATE + n]); s[n] = 0.f; }
  float dpf = Dp[f];
  // prologue: row 0 -> buffer 0
  {
    unsigned r0 = (unsigned)(b << 10) * 128u;
    async_load_b32(lds_base + (unsigned)f * 4u,        (r0 + f) * 4u,      BmCm);
    async_load_b32(lds_base + 256u + (unsigned)f * 4u, (r0 + 64 + f) * 4u, BmCm);
  }
  for (int l = 0; l < LSEQ; ++l) {
    int cur = l & 1;
    if (l + 1 < LSEQ) {
      unsigned rn = (unsigned)((b << 10) + l + 1) * 128u;
      unsigned lb = lds_base + (unsigned)((l + 1) & 1) * 512u;
      async_load_b32(lb + (unsigned)f * 4u,        (rn + f) * 4u,      BmCm);
      async_load_b32(lb + 256u + (unsigned)f * 4u, (rn + 64 + f) * 4u, BmCm);
      wait_async2();
    } else {
      wait_async0();
    }
    __syncthreads();
    const float* sB = &sBC[cur][0];
    const float* sC = &sBC[cur][64];
    int row = (b << 10) + l;
    float d  = Del[row * 64 + f];
    float hv = hc[row * 64 + f];
    float xh = d * hv;
    float acc = 0.f;
    #pragma unroll
    for (int n = 0; n < NSTATE; ++n) {
      float ab = __expf(-d * a[n]);
      s[n] = ab * s[n] + xh * sB[n];
      acc += s[n] * sC[n];
    }
    y[row * 64 + f] = acc + dpf * hv;
    __syncthreads();   // protects double-buffer reuse (WAR with next issue)
  }
}

// ---------------------------------------------------------------------------
// Output projection, fused: g = y*silu(h0); per-row norm; WMMA GEMM; residual.
// ---------------------------------------------------------------------------
__global__ void gemm_out_kernel(const float* __restrict__ Y,
                                const float* __restrict__ H0,
                                const __bf16* __restrict__ W,    // N-major
                                const float* __restrict__ bias,
                                float* __restrict__ h)   // residual in, result out
{
  __shared__ __align__(16) float  ldsG[128 * 64];
  __shared__ __align__(32) __bf16 ldsA[128 * 64];
  __shared__ __align__(32) __bf16 ldsW[64 * 64];
  __shared__ float  rmean[128], rrstd[128];
  const int t = threadIdx.x;
  const int r0 = blockIdx.x * 128;
  const unsigned ldsw_base = lds_addr_of(ldsW);
  for (int idx = t; idx < 64 * 8; idx += 256)       // async DMA weights
    async_load_b128(ldsw_base + (unsigned)idx * 16u, (unsigned)idx * 16u, W);
  for (int idx = t; idx < 128 * 16; idx += 256) {
    float4 yv = ((const float4*)Y)[r0 * 16 + idx];
    float4 hv = ((const float4*)H0)[r0 * 16 + idx];
    float4 g;
    g.x = yv.x * siluf_(hv.x); g.y = yv.y * siluf_(hv.y);
    g.z = yv.z * siluf_(hv.z); g.w = yv.w * siluf_(hv.w);
    ((float4*)ldsG)[idx] = g;
  }
  __syncthreads();
  if (t < 128) {
    float s = 0.f, ss = 0.f;
    #pragma unroll
    for (int k = 0; k < 64; ++k) { float v = ldsG[t * 64 + k]; s += v; ss += v * v; }
    float m = s / 64.f;
    rmean[t] = m; rrstd[t] = rsqrtf(ss / 64.f - m * m + EPSV);
  }
  __syncthreads();
  for (int idx = t; idx < 128 * 16; idx += 256) {
    int row = idx >> 4;
    float m = rmean[row], r = rrstd[row];
    float4 g = ((const float4*)ldsG)[idx];
    uint2 pk;
    pk.x = pack2bf((g.x - m) * r, (g.y - m) * r);
    pk.y = pack2bf((g.z - m) * r, (g.w - m) * r);
    ((uint2*)ldsA)[idx] = pk;
  }
  wait_async0();
  __syncthreads();

  const int lane = t & 31, wave = t >> 5;
  const int m = lane & 15, kh = (lane >> 4) * 8;
  const int n = lane & 15, ko = (lane >> 4) * 16, mh = (lane >> 4) * 8;

  const __bf16* Abase = &ldsA[(wave * 16 + m) * 64];
  v16bf a0 = load_a_frag(Abase, kh, 0);
  v16bf a1 = load_a_frag(Abase, kh, 32);

  for (int nt = 0; nt < 4; ++nt) {
    v8f c;
    float bv = bias[nt * 16 + n];
    #pragma unroll
    for (int i = 0; i < 8; ++i) c[i] = bv;
    v16bf b0 = load_b_frag(ldsW, nt * 16 + n, ko, 0);
    v16bf b1 = load_b_frag(ldsW, nt * 16 + n, ko, 32);
    c = __builtin_amdgcn_wmma_f32_16x16x32_bf16(false, a0, false, b0, (short)0, c, false, false);
    c = __builtin_amdgcn_wmma_f32_16x16x32_bf16(false, a1, false, b1, (short)0, c, false, false);
    #pragma unroll
    for (int i = 0; i < 8; ++i) {
      int o = (r0 + wave * 16 + mh + i) * 64 + nt * 16 + n;
      h[o] = h[o] + c[i];        // residual, element owned by exactly one lane
    }
  }
}

// ---------------------------------------------------------------------------
// ConvTranspose2d k=stride=8 (unpatch).
// ---------------------------------------------------------------------------
__global__ void unpatch_kernel(const float* __restrict__ h,
                               const float* __restrict__ Wout,
                               const float* __restrict__ bout,
                               float* __restrict__ out)
{
  int tid = blockIdx.x * 256 + threadIdx.x;
  if (tid >= BSZ * 3 * 256 * 256) return;
  int xx = tid & 255;
  int yy = (tid >> 8) & 255;
  int c  = (tid >> 16) % 3;
  int b  = tid / (3 * 65536);
  int i = yy >> 3, p = yy & 7, j = xx >> 3, q = xx & 7;
  int l = i * 32 + j;
  const float* hrow = &h[((b << 10) + l) * 64];
  float s = bout[c];
  #pragma unroll
  for (int f = 0; f < 64; ++f)
    s += hrow[f] * Wout[((f * 3 + c) * 8 + p) * 8 + q];
  out[tid] = s;
}

// ---------------------------------------------------------------------------
// Host launcher
// ---------------------------------------------------------------------------
extern "C" void kernel_launch(void* const* d_in, const int* in_sizes, int n_in,
                              void* d_out, int out_size, void* d_ws, size_t ws_size,
                              hipStream_t stream) {
  (void)in_sizes; (void)n_in; (void)out_size; (void)ws_size;
  const float* x      = (const float*)d_in[0];
  const float* Wpe    = (const float*)d_in[1];
  const float* bpe    = (const float*)d_in[2];
  const float* Wp_in  = (const float*)d_in[3];
  const float* bp_in  = (const float*)d_in[4];
  const float* Wconv  = (const float*)d_in[5];
  const float* bconv  = (const float*)d_in[6];
  const float* A_log  = (const float*)d_in[7];
  const float* Dp     = (const float*)d_in[8];
  const float* WB     = (const float*)d_in[9];
  const float* bB     = (const float*)d_in[10];
  const float* WC     = (const float*)d_in[11];
  const float* bC     = (const float*)d_in[12];
  const float* Wd1    = (const float*)d_in[13];
  const float* bd1    = (const float*)d_in[14];
  const float* Wd2    = (const float*)d_in[15];
  const float* bd2    = (const float*)d_in[16];
  const float* Wp_out = (const float*)d_in[17];
  const float* bp_out = (const float*)d_in[18];
  const float* Wout   = (const float*)d_in[19];
  const float* bout   = (const float*)d_in[20];

  char* base = (char*)d_ws;
  size_t off = 0;
  auto alloc = [&](size_t bytes) -> void* {
    void* p = base + off;
    off = (off + bytes + 255) & ~(size_t)255;
    return p;
  };
  __bf16* wpin_b  = (__bf16*)alloc(NLAYERS * 64 * 64  * sizeof(__bf16));
  __bf16* wconv_b = (__bf16*)alloc(NLAYERS * 3 * 64 * 64 * sizeof(__bf16));
  __bf16* wbc_b   = (__bf16*)alloc(NLAYERS * 128 * 64 * sizeof(__bf16));
  __bf16* wpout_b = (__bf16*)alloc(NLAYERS * 64 * 64  * sizeof(__bf16));
  float*  bbc     = (float*) alloc(NLAYERS * 128 * sizeof(float));
  float*  h       = (float*) alloc((size_t)ROWS * 64  * sizeof(float));
  float*  h0      = (float*) alloc((size_t)ROWS * 64  * sizeof(float));
  float*  hcr     = (float*) alloc((size_t)ROWS * 64  * sizeof(float));
  float*  hc      = (float*) alloc((size_t)ROWS * 64  * sizeof(float));
  float*  BmCm    = (float*) alloc((size_t)ROWS * 128 * sizeof(float));
  float*  Del     = (float*) alloc((size_t)ROWS * 64  * sizeof(float));
  float*  yb      = (float*) alloc((size_t)ROWS * 64  * sizeof(float));
  float*  stats   = (float*) alloc(BSZ * 2 * sizeof(float));
  float*  istats  = (float*) alloc(BSZ * 64 * 2 * sizeof(float));

  convert_weights_kernel<<<384, 256, 0, stream>>>(
      Wp_in, Wconv, WB, WC, bB, bC, Wp_out,
      wpin_b, wconv_b, wbc_b, bbc, wpout_b);

  patch_embed_kernel<<<ROWS, 64, 0, stream>>>(x, Wpe, bpe, h);

  for (int i = 0; i < NLAYERS; ++i) {
    gn_stats_kernel<<<BSZ, 256, 0, stream>>>(h, stats);
    gemm_gen_kernel<<<32, 256, 0, stream>>>(h, stats, wpin_b + i * 4096,
                                            bp_in + i * 64, h0, 64);
    conv_gemm_kernel<<<32, 256, 0, stream>>>(h0, wconv_b + i * 3 * 4096,
                                             bconv + i * 64, hcr);
    inorm_stats_kernel<<<BSZ * 64, 256, 0, stream>>>(hcr, istats);
    inorm_apply_kernel<<<(ROWS * 64) / 256, 256, 0, stream>>>(hcr, istats, hc);
    gemm_gen_kernel<<<32, 256, 0, stream>>>(hc, nullptr, wbc_b + i * 8192,
                                            bbc + i * 128, BmCm, 128);
    delta_kernel<<<ROWS, 64, 0, stream>>>(hc, Wd1 + i * 64 * NDELTA, bd1 + i * NDELTA,
                                          Wd2 + i * NDELTA * 64, bd2 + i * 64, Del);
    scan_kernel<<<BSZ, 64, 0, stream>>>(Del, hc, BmCm,
                                        A_log + i * 64 * 64, Dp + i * 64, yb);
    gemm_out_kernel<<<32, 256, 0, stream>>>(yb, h0, wpout_b + i * 4096,
                                            bp_out + i * 64, h);
  }

  unpatch_kernel<<<(BSZ * 3 * 256 * 256) / 256, 256, 0, stream>>>(h, Wout, bout,
                                                                  (float*)d_out);
}
